// IndexerOp_85444079386736
// MI455X (gfx1250) — matrix operand order repaired and checked by reading, not compiled, not run on hardware
//
#include <hip/hip_runtime.h>
#include <hip/hip_bf16.h>

typedef __attribute__((ext_vector_type(16))) __bf16 v16bf;
typedef __attribute__((ext_vector_type(2)))  __bf16 v2bf;
typedef __attribute__((ext_vector_type(8)))  float  v8f;

#define T_LEN  8192
#define NHEADS 64
#define HD     128
#define WAVES  8

// pack two f32 -> two bf16 (RNE) in one 32-bit lane value; clang lowers the
// fptrunc to gfx1250's native bf16 convert path.
__device__ __forceinline__ unsigned packbf2(float a, float b) {
    v2bf p; p[0] = (__bf16)a; p[1] = (__bf16)b;
    unsigned u; __builtin_memcpy(&u, &p, 4); return u;
}

union V16Cast { struct { uint4 lo, hi; } u; v16bf v; };

__global__ void init_pos_kernel(int* rope_pos) {
    int i = blockIdx.x * blockDim.x + threadIdx.x;
    if (i < T_LEN) rope_pos[i] = -1;
}

__global__ void scatter_pos_kernel(const int* __restrict__ lid,
                                   const int* __restrict__ gid,
                                   const int* __restrict__ positions,
                                   int* __restrict__ rope_pos, int L) {
    int j = blockIdx.x * blockDim.x + threadIdx.x;
    if (j < L) rope_pos[lid[j]] = positions[gid[j]];
}

// One wave per 16-row tile: stage->LDS, RoPE (cols 0..63), Y = X*H via WMMA
// computed as D = H(16x32) x X^T(32x16)  (H symmetric), coalesced b128 stores.
__global__ __launch_bounds__(256) void fwht_rope_kernel(
    const __bf16* __restrict__ x, __bf16* __restrict__ y,
    const float* __restrict__ cs, const int* __restrict__ rope_pos,
    int nTiles, int tShift)
{
    __shared__ __align__(16) unsigned int Hlds[128 * 64];        // 32KB bf16 H (packed pairs)
    __shared__ __align__(16) __bf16       Xlds[WAVES][16 * HD];  // 32KB tiles

    // Build H128: h[i][j] = parity(popc(i&j)) ? -1 : +1 (bf16)
    for (int e = threadIdx.x; e < 128 * 64; e += blockDim.x) {
        int i = e >> 6;
        int j = (e & 63) << 1;
        unsigned s0 = (__popc(i & j)       & 1) ? 0xBF80u : 0x3F80u;
        unsigned s1 = (__popc(i & (j | 1)) & 1) ? 0xBF80u : 0x3F80u;
        Hlds[e] = s0 | (s1 << 16);
    }
    __syncthreads();

    const int lane = threadIdx.x & 31;
    const int wave = threadIdx.x >> 5;
    __bf16* tile = Xlds[wave];
    uint4* tile16 = (uint4*)tile;
    const __bf16* Hs = (const __bf16*)Hlds;

    const int gwave  = blockIdx.x * WAVES + wave;
    const int nwaves = gridDim.x * WAVES;
    const float scale = 0.08838834764831845f;  // 128^-0.5 applied in f32

    for (int tIdx = gwave; tIdx < nTiles; tIdx += nwaves) {
        const long long row0 = (long long)tIdx * 16;

        // ---- stage 16 rows x 128 bf16 (4KB contiguous) into LDS ----
        const uint4* src = (const uint4*)(x + row0 * HD);
        #pragma unroll
        for (int c = 0; c < 8; ++c)
            tile16[c * 32 + lane] = src[c * 32 + lane];

        if (tIdx + nwaves < nTiles)  // global_prefetch_b8 for next tile
            __builtin_prefetch(x + (row0 + (long long)nwaves * 16) * HD + lane * 64, 0, 1);

        asm volatile("s_wait_dscnt 0x0" ::: "memory");

        // ---- RoPE (NeoX) on cols [0,64): lane owns pair (lane, lane+32) ----
        #pragma unroll 4
        for (int r = 0; r < 16; ++r) {
            int pos = rope_pos[(int)((row0 + r) >> tShift)];  // wave-uniform
            if (pos >= 0) {
                float c  = (float)(__bf16)cs[pos * 64 + lane];       // bf16-round like ref
                float s  = (float)(__bf16)cs[pos * 64 + 32 + lane];
                float x1 = (float)tile[r * HD + lane];
                float x2 = (float)tile[r * HD + 32 + lane];
                tile[r * HD + lane]      = (__bf16)(x1 * c - x2 * s);
                tile[r * HD + 32 + lane] = (__bf16)(x2 * c + x1 * s);
            }
        }
        asm volatile("s_wait_dscnt 0x0" ::: "memory");

        // ---- B = X^T chunks (32x16 bf16): lane holds N=lane&15, K=8*(lane>>4)+j (j<8), 16+8*(lane>>4)+(j-8) (j>=8) ----
        V16Cast B[4];
        const int brow = (lane & 15) * HD;
        const int bk   = (lane >> 4) * 8;
        #pragma unroll
        for (int kb = 0; kb < 4; ++kb) {
            const uint4* p = (const uint4*)&tile[brow + kb * 32 + bk];
            B[kb].u.lo = p[0];
            B[kb].u.hi = p[2];
        }

        // ---- D(mb) = sum_kb  H_chunk(16x32) x B[kb] ----
        const int am = lane & 15;          // A: M = lane&15
        const int ak = (lane >> 4) * 8;    // A: K base = 8*(lane>=16)
        #pragma unroll
        for (int mb = 0; mb < 8; ++mb) {
            v8f acc = {};
            #pragma unroll
            for (int kb = 0; kb < 4; ++kb) {
                V16Cast A;
                const uint4* hp = (const uint4*)&Hs[(mb * 16 + am) * HD + kb * 32 + ak];
                A.u.lo = hp[0];     // K = ak .. ak+7
                A.u.hi = hp[2];     // K = 16+ak .. 16+ak+7  (+32 bytes)
                acc = __builtin_amdgcn_wmma_f32_16x16x32_bf16(
                        false, A.v, false, B[kb].v, (short)0, acc, false, false);
            }
            // lane holds output row (row0+am), cols mb*16 + 8*(lane>>4) + [0..7]
            uint4 outv;
            outv.x = packbf2(acc[0] * scale, acc[1] * scale);
            outv.y = packbf2(acc[2] * scale, acc[3] * scale);
            outv.z = packbf2(acc[4] * scale, acc[5] * scale);
            outv.w = packbf2(acc[6] * scale, acc[7] * scale);
            const int col0 = mb * 16 + (lane >> 4) * 8;
            *(uint4*)(y + (row0 + am) * HD + col0) = outv;
        }
    }
}

extern "C" void kernel_launch(void* const* d_in, const int* in_sizes, int n_in,
                              void* d_out, int out_size, void* d_ws, size_t ws_size,
                              hipStream_t stream) {
    const __bf16* q   = (const __bf16*)d_in[0];
    const __bf16* k   = (const __bf16*)d_in[1];
    const float*  cs  = (const float*)d_in[2];
    const int*    pos = (const int*)d_in[3];
    const int*    lid = (const int*)d_in[4];
    const int*    gid = (const int*)d_in[5];
    __bf16*       out = (__bf16*)d_out;
    int*     rope_pos = (int*)d_ws;   // T ints = 32KB
    const int L = in_sizes[4];

    init_pos_kernel<<<(T_LEN + 255) / 256, 256, 0, stream>>>(rope_pos);
    scatter_pos_kernel<<<(L + 255) / 256, 256, 0, stream>>>(lid, gid, pos, rope_pos, L);

    const int qTiles = (T_LEN * NHEADS) / 16;  // 32768 tiles
    const int kTiles = T_LEN / 16;             // 512 tiles
    fwht_rope_kernel<<<1024, 256, 0, stream>>>(q, out, cs, rope_pos, qTiles, 6);
    fwht_rope_kernel<<<64, 256, 0, stream>>>(
        k, out + (size_t)T_LEN * NHEADS * HD, cs, rope_pos, kTiles, 0);
}
